// AdmixMultiHeadAttention_56530359550481
// MI455X (gfx1250) — matrix operand order repaired and checked by reading, not compiled, not run on hardware
//
#include <hip/hip_runtime.h>
#include <hip/hip_bf16.h>

// ---------------------------------------------------------------------------
// AdmixMultiHeadAttention fused kernels for MI455X (gfx1250, wave32, WMMA)
//   B=32, S=1024, H=2, DH=64, DE=128, DC=64
//
// Pipeline:
//   proj_kernel : bf16 Q/K projections, transposed bf16 V, padding masks
//   attn_kernel : fused flash attention; per 32-key tile the workgroup
//                 cooperatively stages noise/K/V^T into LDS (double-buffered,
//                 GLOBAL_LOAD_ASYNC_TO_LDS_B128), then each wave runs
//                 12x v_wmma_f32_16x16x32_bf16 + online softmax.
// ---------------------------------------------------------------------------

typedef __attribute__((ext_vector_type(16))) __bf16       v16bf;
typedef __attribute__((ext_vector_type(8)))  float        v8f;
typedef __attribute__((ext_vector_type(4)))  unsigned int u32x4;
typedef int v4i_vs __attribute__((__vector_size__(4 * sizeof(int))));  // GCC vector, matches builtin

constexpr int B  = 32;
constexpr int S  = 1024;
constexpr int DH = 64;
constexpr int DE = 128;
constexpr int DC = 64;

// Padded LDS row widths (elements) — keep 16B chunk alignment, break the
// 64/128-byte row strides that would alias LDS banks on b128 gathers.
constexpr int NR = 36;   // noise rows: 32 f32 used   (144 B row)
constexpr int KR = 72;   // K rows:     64 bf16 used  (144 B row)
constexpr int VR = 40;   // V^T rows:   32 bf16 used  ( 80 B row)
constexpr int PR = 40;   // P rows:     32 bf16 used  ( 80 B row)

union FragU { v16bf v; u32x4 u[2]; };

__device__ __forceinline__ v16bf load_frag16(const __bf16* p0, const __bf16* p1) {
  FragU f;
  f.u[0] = *reinterpret_cast<const u32x4*>(p0);
  f.u[1] = *reinterpret_cast<const u32x4*>(p1);
  return f.v;
}

__device__ __forceinline__ float hmax16(float v) {
#pragma unroll
  for (int m = 1; m <= 8; m <<= 1) v = fmaxf(v, __shfl_xor(v, m, 32));
  return v;
}
__device__ __forceinline__ float hsum16(float v) {
#pragma unroll
  for (int m = 1; m <= 8; m <<= 1) v += __shfl_xor(v, m, 32);
  return v;
}

// Same-wave LDS store->load fence (split dscnt counter on CDNA5).
__device__ __forceinline__ void lds_wave_fence() {
  __builtin_amdgcn_wave_barrier();
  asm volatile("s_wait_dscnt 0" ::: "memory");
  __builtin_amdgcn_wave_barrier();
}

// ---- async global->LDS copy (CDNA5 GLOBAL_LOAD_ASYNC_TO_LDS_B128) ---------
#if __has_builtin(__builtin_amdgcn_global_load_async_to_lds_b128)
#define HAS_ASYNC_LDS 1
#else
#define HAS_ASYNC_LDS 0
#endif

__device__ __forceinline__ void async_copy_b128(const void* g, void* l) {
#if HAS_ASYNC_LDS
  __builtin_amdgcn_global_load_async_to_lds_b128(
      (__attribute__((address_space(1))) v4i_vs*)(g),
      (__attribute__((address_space(3))) v4i_vs*)(l),
      /*offset=*/0, /*cpol=*/0);
#else
  *reinterpret_cast<u32x4*>(l) = *reinterpret_cast<const u32x4*>(g);
#endif
}

template <int N>
__device__ __forceinline__ void async_wait() {
#if HAS_ASYNC_LDS
  asm volatile("s_wait_asynccnt %0" :: "i"(N) : "memory");
#endif
}

// ---------------------------------------------------------------------------
// Kernel 1: projections + masks.
// ---------------------------------------------------------------------------
__global__ __launch_bounds__(256)
void proj_kernel(const float* __restrict__ queries_it, const float* __restrict__ queries_ctx,
                 const float* __restrict__ keys_it,    const float* __restrict__ keys_ctx,
                 const float* __restrict__ Wq_it, const float* __restrict__ bq_it,
                 const float* __restrict__ Wk_it, const float* __restrict__ bk_it,
                 const float* __restrict__ Wq_ctx, const float* __restrict__ bq_ctx,
                 const float* __restrict__ Wk_ctx, const float* __restrict__ bk_ctx,
                 const float* __restrict__ Wv, const float* __restrict__ bv,
                 __bf16* __restrict__ Qb_it, __bf16* __restrict__ Qb_ctx,
                 __bf16* __restrict__ Kb_it, __bf16* __restrict__ Kb_ctx,
                 __bf16* __restrict__ Vt,
                 float* __restrict__ kmask, float* __restrict__ qmask)
{
  __shared__ float sQ[16][DE];
  __shared__ float sK[16][DE];
  __shared__ float sQc[16][DC];
  __shared__ float sKc[16][DC];

  const int b  = blockIdx.y;
  const int s0 = blockIdx.x * 16;
  const int t  = threadIdx.x;

  for (int i = t; i < 16 * DE; i += 256) {
    const int r = i >> 7, c = i & (DE - 1);
    sQ[r][c] = queries_it[(size_t)(b * S + s0 + r) * DE + c];
    sK[r][c] = keys_it   [(size_t)(b * S + s0 + r) * DE + c];
  }
  for (int i = t; i < 16 * DC; i += 256) {
    const int r = i >> 6, c = i & (DC - 1);
    sQc[r][c] = queries_ctx[(size_t)(b * S + s0 + r) * DC + c];
    sKc[r][c] = keys_ctx   [(size_t)(b * S + s0 + r) * DC + c];
  }
  __syncthreads();

  if (t < 16) {
    float a = 0.f;
    for (int c = 0; c < DE; ++c) a += fabsf(sK[t][c]);
    kmask[b * S + s0 + t] = (a > 0.f) ? 1.f : 0.f;
  } else if (t < 32) {
    const int r = t - 16;
    float a = 0.f;
    for (int c = 0; c < DE; ++c) a += fabsf(sQ[r][c]);
    qmask[b * S + s0 + r] = (a > 0.f) ? 1.f : 0.f;
  }

  const int r = t >> 4;
  const int g = t & 15;

#pragma unroll
  for (int dd = 0; dd < 4; ++dd) {
    const int d = g * 4 + dd;
    float aq = bq_it[d], ak = bk_it[d];
#pragma unroll 4
    for (int c = 0; c < DE; ++c) {
      aq = fmaf(sQ[r][c], Wq_it[d * DE + c], aq);
      ak = fmaf(sK[r][c], Wk_it[d * DE + c], ak);
    }
    Qb_it[(size_t)(b * S + s0 + r) * DH + d] = (__bf16)aq;
    Kb_it[(size_t)(b * S + s0 + r) * DH + d] = (__bf16)ak;

    float aqc = bq_ctx[d], akc = bk_ctx[d];
#pragma unroll 4
    for (int c = 0; c < DC; ++c) {
      aqc = fmaf(sQc[r][c], Wq_ctx[d * DC + c], aqc);
      akc = fmaf(sKc[r][c], Wk_ctx[d * DC + c], akc);
    }
    Qb_ctx[(size_t)(b * S + s0 + r) * DH + d] = (__bf16)aqc;
    Kb_ctx[(size_t)(b * S + s0 + r) * DH + d] = (__bf16)akc;
  }

#pragma unroll
  for (int dd = 0; dd < 8; ++dd) {
    const int d = g * 8 + dd;
    float av = bv[d];
#pragma unroll 4
    for (int c = 0; c < DE; ++c) av = fmaf(sK[r][c], Wv[d * DE + c], av);
    const int hh = d >> 6, dh = d & 63;
    Vt[((size_t)(b * 2 + hh) * DH + dh) * S + s0 + r] = (__bf16)av;
  }
}

// ---------------------------------------------------------------------------
// Stage one 32-key tile (noise ch0/ch1, K_it, K_ctx, V^T head0/head1) into
// LDS.  128 threads x 12 b128 chunks each (ASYNCcnt-tracked).
// ---------------------------------------------------------------------------
__device__ __forceinline__ void issue_tile(int b, int q0, int kt, int t,
                                           const float* __restrict__ noise,
                                           const __bf16* __restrict__ Kb_it,
                                           const __bf16* __restrict__ Kb_ctx,
                                           const __bf16* __restrict__ Vt,
                                           float* nb, __bf16* kb, __bf16* vb)
{
  // noise: [2 ch][32 rows][8 x 16B chunks]  (rows q0..q0+31, cols kt..kt+31)
#pragma unroll
  for (int i = 0; i < 4; ++i) {
    const int idx = t + i * 128;           // 0..511
    const int ch  = idx >> 8;
    const int row = (idx >> 3) & 31;
    const int ck  = idx & 7;
    const float* g = noise + ((size_t)(b * 2 + ch) * S + q0 + row) * S + kt + ck * 4;
    async_copy_b128(g, nb + (ch * 32 + row) * NR + ck * 4);
  }
  // K: [2 mats][32 keys][8 x 16B chunks]
#pragma unroll
  for (int i = 0; i < 4; ++i) {
    const int idx = t + i * 128;
    const int mt  = idx >> 8;
    const int key = (idx >> 3) & 31;
    const int ck  = idx & 7;
    const __bf16* src = (mt ? Kb_ctx : Kb_it) + (size_t)(b * S + kt + key) * DH + ck * 8;
    async_copy_b128(src, kb + (mt * 32 + key) * KR + ck * 8);
  }
  // V^T: [2 heads][64 d-rows][4 x 16B chunks]
#pragma unroll
  for (int i = 0; i < 4; ++i) {
    const int idx = t + i * 128;
    const int hh  = idx >> 8;
    const int d   = (idx >> 2) & 63;
    const int ck  = idx & 3;
    const __bf16* src = Vt + ((size_t)(b * 2 + hh) * DH + d) * S + kt + ck * 8;
    async_copy_b128(src, vb + (hh * 64 + d) * VR + ck * 8);
  }
}

// ---------------------------------------------------------------------------
// Kernel 2: fused flash attention.  Grid (S/32, B), 128 threads = 4 waves;
// wave = (head h, q-subtile qs).  Double-buffered LDS tiles.
// ---------------------------------------------------------------------------
__global__ __launch_bounds__(128)
void attn_kernel(const float* __restrict__ queries_it,
                 const float* __restrict__ noise,
                 const float* __restrict__ sigma_noise,
                 const float* __restrict__ W1, const float* __restrict__ b1,
                 const float* __restrict__ W2, const float* __restrict__ b2,
                 const __bf16* __restrict__ Qb_it, const __bf16* __restrict__ Qb_ctx,
                 const __bf16* __restrict__ Kb_it, const __bf16* __restrict__ Kb_ctx,
                 const __bf16* __restrict__ Vt,
                 const float* __restrict__ kmask, const float* __restrict__ qmask,
                 float* __restrict__ out)
{
  __shared__ float  nbuf[2][2 * 32 * NR];   // ~18 KB
  __shared__ __bf16 kbuf[2][2 * 32 * KR];   // ~18 KB
  __shared__ __bf16 vbuf[2][2 * 64 * VR];   // ~20 KB
  __shared__ __bf16 pstage[4][16 * PR];     //  ~5 KB

  const int b     = blockIdx.y;
  const int q0    = blockIdx.x * 32;
  const int tid   = threadIdx.x;
  const int wave  = tid >> 5;
  const int lane  = tid & 31;
  const int h     = wave >> 1;            // head 0/1
  const int qs    = wave & 1;             // q sub-tile
  const int half  = lane >> 4;
  const int ln    = lane & 15;
  const int qrow0 = q0 + qs * 16;

  // Uniform tiny-MLP / noise parameters.
  const float w100 = W1[0], w101 = W1[1], w110 = W1[2], w111 = W1[3];
  const float b10 = b1[0], b11 = b1[1];
  const float w2h0 = W2[h * 2 + 0], w2h1 = W2[h * 2 + 1], b2h = b2[h];
  float sg0 = sigma_noise[b * 2 + 0]; sg0 *= sg0;
  float sg1 = sigma_noise[b * 2 + 1]; sg1 *= sg1;

  // Q A-fragments (ISA 16-bit A layout: lane=row, elems {halfK..halfK+7, +16}).
  const __bf16* qi = Qb_it  + (size_t)(b * S + qrow0 + ln) * DH;
  const __bf16* qc = Qb_ctx + (size_t)(b * S + qrow0 + ln) * DH;
  v16bf aqit[2], aqctx[2];
#pragma unroll
  for (int c = 0; c < 2; ++c) {
    aqit[c]  = load_frag16(qi + c * 32 + half * 8, qi + c * 32 + 16 + half * 8);
    aqctx[c] = load_frag16(qc + c * 32 + half * 8, qc + c * 32 + 16 + half * 8);
  }

  v8f o[4] = {};
  float m_run[8], l_run[8];
#pragma unroll
  for (int r = 0; r < 8; ++r) { m_run[r] = -3.0e38f; l_run[r] = 0.0f; }

  __bf16* pt = &pstage[wave][0];

  // Prologue: stage first tile into buffer 0.
  issue_tile(b, q0, 0, tid, noise, Kb_it, Kb_ctx, Vt,
             nbuf[0], kbuf[0], vbuf[0]);
  int cur = 0;

  for (int kt = 0; kt < S; kt += 32) {
    const bool hasnext = (kt + 32) < S;
    if (hasnext) {
      issue_tile(b, q0, kt + 32, tid, noise, Kb_it, Kb_ctx, Vt,
                 nbuf[cur ^ 1], kbuf[cur ^ 1], vbuf[cur ^ 1]);
      async_wait<12>();                 // current tile (issued last iter) done
    } else {
      async_wait<0>();
    }
    __syncthreads();                    // all waves' shares visible

    const float*  nb = nbuf[cur];
    const __bf16* kb = kbuf[cur];
    const __bf16* vb = vbuf[cur];

    // ---- dual-channel scores for two 16-key subtiles -------------------
    v8f sit[2], sctx[2];
    float km[2];
#pragma unroll
    for (int sub = 0; sub < 2; ++sub) {
      const int kl = sub * 16 + ln;     // key within tile
      km[sub] = kmask[b * S + kt + kl];
      const __bf16* kbi = kb + (0 * 32 + kl) * KR;
      const __bf16* kbc = kb + (1 * 32 + kl) * KR;
      v8f zi = {}, zc = {};
#pragma unroll
      for (int c = 0; c < 2; ++c) {     // split-K over DH=64
        v16bf bk = load_frag16(kbi + c * 32 + half * 8, kbi + c * 32 + 16 + half * 8);
        zi = __builtin_amdgcn_wmma_f32_16x16x32_bf16(false, aqit[c],  false, bk,
                                                     (short)0, zi, false, false);
        v16bf bc = load_frag16(kbc + c * 32 + half * 8, kbc + c * 32 + 16 + half * 8);
        zc = __builtin_amdgcn_wmma_f32_16x16x32_bf16(false, aqctx[c], false, bc,
                                                     (short)0, zc, false, false);
      }
      sit[sub] = zi; sctx[sub] = zc;
    }

    // ---- noise + per-element HDP MLP + key mask (C layout) -------------
    float pv[2][8];
#pragma unroll
    for (int sub = 0; sub < 2; ++sub) {
      const int cloc = sub * 16 + ln;
#pragma unroll
      for (int r = 0; r < 8; ++r) {
        const int rloc = qs * 16 + r + half * 8;   // row within 32-row tile
        const float x0 = sit[sub][r]  + nb[(0 * 32 + rloc) * NR + cloc] * sg0;
        const float x1 = sctx[sub][r] + nb[(1 * 32 + rloc) * NR + cloc] * sg1;
        const float h0 = fmaxf(0.f, fmaf(w100, x0, fmaf(w101, x1, b10)));
        const float h1 = fmaxf(0.f, fmaf(w110, x0, fmaf(w111, x1, b11)));
        const float sv = fmaf(w2h0, h0, fmaf(w2h1, h1, b2h)) * 0.125f;
        pv[sub][r] = (km[sub] == 0.0f) ? -1.0e9f : sv;
      }
    }

    // ---- online softmax update ----------------------------------------
    float scale[8];
#pragma unroll
    for (int r = 0; r < 8; ++r) {
      const float rowmax = hmax16(fmaxf(pv[0][r], pv[1][r]));
      const float mnew   = fmaxf(m_run[r], rowmax);
      scale[r] = __expf(m_run[r] - mnew);
      const float p0 = __expf(pv[0][r] - mnew);
      const float p1 = __expf(pv[1][r] - mnew);
      l_run[r] = l_run[r] * scale[r] + hsum16(p0 + p1);
      m_run[r] = mnew;
      pv[0][r] = p0; pv[1][r] = p1;
    }
#pragma unroll
    for (int d = 0; d < 4; ++d)
#pragma unroll
      for (int r = 0; r < 8; ++r) o[d][r] *= scale[r];

    // ---- stage P to LDS (row-major [16 q][PR] bf16), reload as A-frag --
#pragma unroll
    for (int sub = 0; sub < 2; ++sub)
#pragma unroll
      for (int r = 0; r < 8; ++r)
        pt[(r + half * 8) * PR + sub * 16 + ln] = (__bf16)pv[sub][r];
    lds_wave_fence();
    const v16bf pa = load_frag16(pt + ln * PR + half * 8,
                                 pt + ln * PR + 16 + half * 8);

    // ---- O += P @ V^T --------------------------------------------------
#pragma unroll
    for (int d = 0; d < 4; ++d) {
      const __bf16* vp = vb + (h * 64 + d * 16 + ln) * VR + half * 8;
      const v16bf vbf = load_frag16(vp, vp + 16);
      o[d] = __builtin_amdgcn_wmma_f32_16x16x32_bf16(false, pa, false, vbf,
                                                     (short)0, o[d], false, false);
    }

    __syncthreads();                    // all reads of cur done before reuse
    cur ^= 1;
  }

  // ---- epilogue: 1/l, query mask, residual add, store ------------------
  float inv[8], qm[8];
#pragma unroll
  for (int r = 0; r < 8; ++r) {
    inv[r] = 1.0f / l_run[r];
    qm[r]  = qmask[b * S + qrow0 + r + half * 8];
  }
#pragma unroll
  for (int d = 0; d < 4; ++d) {
#pragma unroll
    for (int r = 0; r < 8; ++r) {
      const int row = qrow0 + r + half * 8;
      const int col = h * 64 + d * 16 + ln;
      const size_t idx = (size_t)(b * S + row) * DE + col;
      out[idx] = o[d][r] * inv[r] * qm[r] + queries_it[idx];
    }
  }
}

// ---------------------------------------------------------------------------
extern "C" void kernel_launch(void* const* d_in, const int* in_sizes, int n_in,
                              void* d_out, int out_size, void* d_ws, size_t ws_size,
                              hipStream_t stream) {
  (void)in_sizes; (void)n_in; (void)out_size; (void)ws_size;

  const float* queries_it  = (const float*)d_in[0];
  const float* queries_ctx = (const float*)d_in[1];
  const float* keys_it     = (const float*)d_in[2];
  const float* keys_ctx    = (const float*)d_in[3];
  const float* sigma_noise = (const float*)d_in[4];
  const float* noise       = (const float*)d_in[5];
  const float* Wq_it  = (const float*)d_in[6];  const float* bq_it  = (const float*)d_in[7];
  const float* Wk_it  = (const float*)d_in[8];  const float* bk_it  = (const float*)d_in[9];
  const float* Wq_ctx = (const float*)d_in[10]; const float* bq_ctx = (const float*)d_in[11];
  const float* Wk_ctx = (const float*)d_in[12]; const float* bk_ctx = (const float*)d_in[13];
  const float* Wv     = (const float*)d_in[14]; const float* bv     = (const float*)d_in[15];
  const float* W1     = (const float*)d_in[16]; const float* b1     = (const float*)d_in[17];
  const float* W2     = (const float*)d_in[18]; const float* b2     = (const float*)d_in[19];
  float* out = (float*)d_out;

  // Workspace layout (~24.3 MB total).
  char* ws = (char*)d_ws;
  size_t off = 0;
  auto alloc = [&](size_t bytes) {
    char* p = ws + off;
    off += (bytes + 255) & ~(size_t)255;
    return p;
  };
  const size_t qk_bytes = (size_t)B * S * DH * sizeof(__bf16);
  __bf16* Qb_it  = (__bf16*)alloc(qk_bytes);
  __bf16* Qb_ctx = (__bf16*)alloc(qk_bytes);
  __bf16* Kb_it  = (__bf16*)alloc(qk_bytes);
  __bf16* Kb_ctx = (__bf16*)alloc(qk_bytes);
  __bf16* Vt     = (__bf16*)alloc((size_t)B * 2 * DH * S * sizeof(__bf16));
  float*  kmask  = (float*)alloc((size_t)B * S * sizeof(float));
  float*  qmask  = (float*)alloc((size_t)B * S * sizeof(float));

  dim3 gp(S / 16, B);
  proj_kernel<<<gp, 256, 0, stream>>>(queries_it, queries_ctx, keys_it, keys_ctx,
                                      Wq_it, bq_it, Wk_it, bk_it,
                                      Wq_ctx, bq_ctx, Wk_ctx, bk_ctx,
                                      Wv, bv,
                                      Qb_it, Qb_ctx, Kb_it, Kb_ctx, Vt,
                                      kmask, qmask);

  dim3 ga(S / 32, B);
  attn_kernel<<<ga, 128, 0, stream>>>(queries_it, noise, sigma_noise,
                                      W1, b1, W2, b2,
                                      Qb_it, Qb_ctx, Kb_it, Kb_ctx, Vt,
                                      kmask, qmask, out);
}
